// GNN_21337397527225
// MI455X (gfx1250) — matrix operand order repaired and checked by reading, not compiled, not run on hardware
//
#include <hip/hip_runtime.h>

constexpr int N_NODES = 100000;
constexpr int N_EDGES = 1600000;
constexpr int DF      = 128;

typedef __attribute__((ext_vector_type(16))) _Float16 v16h;
typedef __attribute__((ext_vector_type(8)))  _Float16 v8h;
typedef __attribute__((ext_vector_type(4)))  _Float16 v4h;
typedef __attribute__((ext_vector_type(8)))  float    v8f;
typedef __attribute__((ext_vector_type(4)))  float    v4f;

// ---------------------------------------------------------------- utilities

__global__ void zero_f32_v4(float* __restrict__ p, int n4) {
  int i = blockIdx.x * blockDim.x + threadIdx.x;
  if (i < n4) {
    v4f z = {0.0f, 0.0f, 0.0f, 0.0f};
    *(v4f*)(p + i * 4) = z;
  }
}

__global__ void cvt_f32_f16_v4(const float* __restrict__ s,
                               _Float16* __restrict__ d, int n4) {
  int i = blockIdx.x * blockDim.x + threadIdx.x;
  if (i < n4) {
    v4f v = *(const v4f*)(s + i * 4);
    v4h h = {(_Float16)v[0], (_Float16)v[1], (_Float16)v[2], (_Float16)v[3]};
    *(v4h*)(d + i * 4) = h;
  }
}

// ------------------------------------------------------------- aggregation

__global__ void deg_count(const long long* __restrict__ dst,
                          float* __restrict__ deg, int nedges) {
  int e = blockIdx.x * blockDim.x + threadIdx.x;
  if (e < nedges) {
    unsafeAtomicAdd(&deg[(int)dst[e]], 1.0f);   // global_atomic_add_f32
  }
}

// one wave per edge: lane L handles features [4L, 4L+4) (8B coalesced gather,
// 4 fp32 atomic adds into the L2-resident accumulator)
__global__ void scatter_accum(const _Float16* __restrict__ feat,
                              const long long* __restrict__ src,
                              const long long* __restrict__ dst,
                              float* __restrict__ agg, int nedges) {
  int tid  = blockIdx.x * blockDim.x + threadIdx.x;
  int e    = tid >> 5;
  int lane = tid & 31;
  if (e >= nedges) return;
  int s = (int)src[e];
  int d = (int)dst[e];
  v4h hv = *(const v4h*)(feat + s * DF + lane * 4);
  float* o = agg + d * DF + lane * 4;
  unsafeAtomicAdd(o + 0, (float)hv[0]);
  unsafeAtomicAdd(o + 1, (float)hv[1]);
  unsafeAtomicAdd(o + 2, (float)hv[2]);
  unsafeAtomicAdd(o + 3, (float)hv[3]);
}

// agg / max(deg,1) -> f16 (A-matrix input for WMMA)
__global__ void norm_cvt(const float* __restrict__ agg,
                         const float* __restrict__ deg,
                         _Float16* __restrict__ out, int n4) {
  int i = blockIdx.x * blockDim.x + threadIdx.x;
  if (i >= n4) return;
  int i4 = i * 4;                       // 4 consecutive elems share a row
  float invd = 1.0f / fmaxf(deg[i4 >> 7], 1.0f);
  v4f v = *(const v4f*)(agg + i4);
  v4h h = {(_Float16)(v[0] * invd), (_Float16)(v[1] * invd),
           (_Float16)(v[2] * invd), (_Float16)(v[3] * invd)};
  *(v4h*)(out + i4) = h;
}

// ------------------------------------------------------------- fused GEMM
// out[n,j] = sum_k Al[n,k]*Wl[j,k] + b[j] + sum_k Ar[n,k]*Wr[j,k]  (opt ReLU)
// One wave -> 16 rows x 128 cols (8 C tiles). K loop: 4 steps of 32.
__global__ __launch_bounds__(256) void sage_gemm(
    const _Float16* __restrict__ Al,   // [N,128] f16 (normalized aggregate)
    const _Float16* __restrict__ Ar,   // [N,128] f16 (root features)
    const _Float16* __restrict__ Wl,   // [128,128] f16, row j = output col j
    const _Float16* __restrict__ Wr,   // [128,128] f16
    const float*    __restrict__ bias, // [128] f32
    float*          __restrict__ out_f32,  // may be null
    _Float16*       __restrict__ out_f16,  // may be null
    int nrows, int do_relu) {
  const int lane    = threadIdx.x & 31;
  const int wave    = threadIdx.x >> 5;
  const int rowbase = (blockIdx.x * 8 + wave) * 16;
  const int r16     = lane & 15;
  const int hk      = lane >> 4;

  // clamp A row for loads (EXEC must stay all-ones around WMMA); stores guarded
  int arow = rowbase + r16;
  if (arow >= nrows) arow = nrows - 1;
  const _Float16* alp = Al + arow * DF;
  const _Float16* arp = Ar + arow * DF;
  const _Float16* wlp = Wl + r16 * DF;   // + ct*16*DF selects the col tile
  const _Float16* wrp = Wr + r16 * DF;

  v8f acc[8];
  v8f zero = {};
#pragma unroll
  for (int i = 0; i < 8; ++i) acc[i] = zero;

#pragma unroll
  for (int ks = 0; ks < 4; ++ks) {
    // A layout: lane holds contiguous k = ks*32 + hk*16 + [0..15]
    int ka = ks * 32 + hk * 16;
    v8h a1lo = *(const v8h*)(alp + ka);
    v8h a1hi = *(const v8h*)(alp + ka + 8);
    v8h a2lo = *(const v8h*)(arp + ka);
    v8h a2hi = *(const v8h*)(arp + ka + 8);
    v16h a1 = __builtin_shufflevector(a1lo, a1hi, 0, 1, 2, 3, 4, 5, 6, 7,
                                      8, 9, 10, 11, 12, 13, 14, 15);
    v16h a2 = __builtin_shufflevector(a2lo, a2hi, 0, 1, 2, 3, 4, 5, 6, 7,
                                      8, 9, 10, 11, 12, 13, 14, 15);
    // B layout: lane holds contiguous k = ks*32 + hk*16 + [0..15] of W row r16
    int kb = ks * 32 + hk * 16;
#pragma unroll
    for (int ct = 0; ct < 8; ++ct) {
      v16h bl = *(const v16h*)(wlp + ct * 16 * DF + kb);
      acc[ct] = __builtin_amdgcn_wmma_f32_16x16x32_f16(
          false, a1, false, bl, (short)0, acc[ct], false, false);
      v16h br = *(const v16h*)(wrp + ct * 16 * DF + kb);
      acc[ct] = __builtin_amdgcn_wmma_f32_16x16x32_f16(
          false, a2, false, br, (short)0, acc[ct], false, false);
    }
  }

  // epilogue: C layout -> row = rowbase + r + (lane>=16 ? 8 : 0), col = ct*16 + r16
  const int mh = hk * 8;
#pragma unroll
  for (int ct = 0; ct < 8; ++ct) {
    int   col = ct * 16 + r16;
    float bv  = bias[col];
#pragma unroll
    for (int r = 0; r < 8; ++r) {
      int row = rowbase + r + mh;
      if (row < nrows) {
        float v = acc[ct][r] + bv;
        if (do_relu) v = fmaxf(v, 0.0f);
        if (out_f32) out_f32[row * DF + col] = v;
        if (out_f16) out_f16[row * DF + col] = (_Float16)v;
      }
    }
  }
}

// ---------------------------------------------------------------- launcher

extern "C" void kernel_launch(void* const* d_in, const int* in_sizes, int n_in,
                              void* d_out, int out_size, void* d_ws, size_t ws_size,
                              hipStream_t stream) {
  const float*     x   = (const float*)d_in[0];
  const long long* ei  = (const long long*)d_in[1];   // int64 [2, E]
  const float*     W1l = (const float*)d_in[2];
  const float*     b1  = (const float*)d_in[3];
  const float*     W1r = (const float*)d_in[4];
  const float*     W2l = (const float*)d_in[5];
  const float*     b2  = (const float*)d_in[6];
  const float*     W2r = (const float*)d_in[7];
  const long long* src = ei;
  const long long* dst = ei + N_EDGES;

  char*  ws  = (char*)d_ws;
  size_t off = 0;
  auto take = [&](size_t bytes) -> void* {
    void* p = ws + off;
    off = (off + bytes + 255) & ~(size_t)255;
    return p;
  };
  float*    deg   = (float*)take((size_t)N_NODES * 4);
  float*    agg   = (float*)take((size_t)N_NODES * DF * 4);
  _Float16* x_h   = (_Float16*)take((size_t)N_NODES * DF * 2);
  _Float16* h_h   = (_Float16*)take((size_t)N_NODES * DF * 2);
  _Float16* an_h  = (_Float16*)take((size_t)N_NODES * DF * 2);
  _Float16* w1l_h = (_Float16*)take((size_t)DF * DF * 2);
  _Float16* w1r_h = (_Float16*)take((size_t)DF * DF * 2);
  _Float16* w2l_h = (_Float16*)take((size_t)DF * DF * 2);
  _Float16* w2r_h = (_Float16*)take((size_t)DF * DF * 2);

  const int B    = 256;
  const int nd4  = N_NODES * DF / 4;         // 3,200,000
  const int w4   = DF * DF / 4;              // 4,096
  const int n4   = N_NODES / 4;              // 25,000
  const int gcta = (nd4 + B - 1) / B;
  const int gctw = (w4 + B - 1) / B;
  const int gemm_blocks = (((N_NODES + 15) / 16) + 7) / 8;   // 782
  const int scat_blocks = (N_EDGES * 32 + B - 1) / B;        // 200,000

  // precision prep: f16 gather/GEMM operands
  cvt_f32_f16_v4<<<gcta, B, 0, stream>>>(x,   x_h,   nd4);
  cvt_f32_f16_v4<<<gctw, B, 0, stream>>>(W1l, w1l_h, w4);
  cvt_f32_f16_v4<<<gctw, B, 0, stream>>>(W1r, w1r_h, w4);
  cvt_f32_f16_v4<<<gctw, B, 0, stream>>>(W2l, w2l_h, w4);
  cvt_f32_f16_v4<<<gctw, B, 0, stream>>>(W2r, w2r_h, w4);

  // degrees (graph is shared by both layers)
  zero_f32_v4<<<(n4 + B - 1) / B, B, 0, stream>>>(deg, n4);
  deg_count<<<(N_EDGES + B - 1) / B, B, 0, stream>>>(dst, deg, N_EDGES);

  // ---- layer 1: h = relu(mean_agg(x) @ W1l^T + b1 + x @ W1r^T) -> f16
  zero_f32_v4<<<gcta, B, 0, stream>>>(agg, nd4);
  scatter_accum<<<scat_blocks, B, 0, stream>>>(x_h, src, dst, agg, N_EDGES);
  norm_cvt<<<gcta, B, 0, stream>>>(agg, deg, an_h, nd4);
  sage_gemm<<<gemm_blocks, B, 0, stream>>>(an_h, x_h, w1l_h, w1r_h, b1,
                                           nullptr, h_h, N_NODES, 1);

  // ---- layer 2: out = mean_agg(h) @ W2l^T + b2 + h @ W2r^T -> fp32 d_out
  zero_f32_v4<<<gcta, B, 0, stream>>>(agg, nd4);
  scatter_accum<<<scat_blocks, B, 0, stream>>>(h_h, src, dst, agg, N_EDGES);
  norm_cvt<<<gcta, B, 0, stream>>>(agg, deg, an_h, nd4);
  sage_gemm<<<gemm_blocks, B, 0, stream>>>(an_h, h_h, w2l_h, w2r_h, b2,
                                           (float*)d_out, nullptr, N_NODES, 0);
}
